// Attention_3874060501037
// MI455X (gfx1250) — matrix operand order repaired and checked by reading, not compiled, not run on hardware
//
#include <hip/hip_runtime.h>
#include <hip/hip_bf16.h>

// ---------------- problem constants ----------------
#define S_LEN   2048
#define HIDDEN  2048
#define HEAD_D  64
#define N_HEADS 32
#define N_KV    8
#define QKVO    ((N_HEADS + 2 * N_KV) * HEAD_D)   // 3072
#define AO_COLS (N_HEADS * HEAD_D)                // 2048

// ---------------- WMMA fragment types ----------------
typedef __attribute__((ext_vector_type(16))) __bf16 v16bf;
typedef __attribute__((ext_vector_type(8)))  float  v8f;
typedef __attribute__((ext_vector_type(4)))  unsigned int v4u;

union Frag { v4u q[2]; v16bf v; };

__device__ __forceinline__ unsigned short f2bf(float f) {
  unsigned int u = __float_as_uint(f);
  u += 0x7FFFu + ((u >> 16) & 1u);        // round-to-nearest-even
  return (unsigned short)(u >> 16);
}

// A-fragment (16x32 bf16, MxK): lane(0..15) row=lane, K=0..15 contiguous;
// lane(16..31) row=lane-16, K=16..31 contiguous. Row is K-contiguous in memory.
__device__ __forceinline__ v16bf load_a_frag(const unsigned short* __restrict__ A,
                                             int row, int ld, int k0, int lane) {
  const unsigned short* p = A + (size_t)row * ld + k0 + ((lane >> 4) << 4);
  Frag f;
  f.q[0] = *(const v4u*)p;
  f.q[1] = *(const v4u*)(p + 8);
  return f.v;
}

// B-fragment (32x16 bf16, KxN): lane(0..15) col=lane, K=0..15 contiguous;
// lane(16..31) col=lane-16, K=16..31 contiguous. Column is K-contiguous in memory.
__device__ __forceinline__ v16bf load_b_frag(const unsigned short* __restrict__ B,
                                             int col, int ld, int k0, int lane) {
  const unsigned short* p = B + (size_t)col * ld + k0 + ((lane >> 4) << 4);
  Frag f;
  f.q[0] = *(const v4u*)p;
  f.q[1] = *(const v4u*)(p + 8);
  return f.v;
}

#define WMMA_BF16(a, b, c) \
  __builtin_amdgcn_wmma_f32_16x16x32_bf16(false, (a), false, (b), (short)0, (c), false, false)

// ---------------- fp32 -> bf16 conversion ----------------
__global__ void cvt_bf16(const float* __restrict__ in, unsigned short* __restrict__ out, int n) {
  int i = blockIdx.x * blockDim.x + threadIdx.x;
  int stride = gridDim.x * blockDim.x;
  for (; i < n; i += stride) out[i] = f2bf(in[i]);
}

// ---------------- shared GEMM mainloop ----------------
// One wave computes a 64x64 tile of C = A[M,K] * B[N,K]^T.
// The 64x32 B-slab per K-step is staged once into LDS through the CDNA5
// async global->LDS path (ASYNCcnt), eliminating the 4x per-half-wave
// duplication of direct fragment loads. A stays direct-from-global (its
// fragment pattern is already 1 b128 per lane).
#define BS_STRIDE 40   // ushort elems per staged B row (80B: 16B-aligned, bank-skewed)

__device__ __forceinline__ void gemm_mainloop(const unsigned short* __restrict__ A,
                                              const unsigned short* __restrict__ B,
                                              unsigned short* Bs,  // LDS, 64*BS_STRIDE
                                              int m0, int n0, int K, int lane,
                                              v8f acc[4][4]) {
  const int lc = lane & 15;

  // lane assignments for the 64x32 bf16 (4KB) B tile: 8 x 16B chunks per lane
  const unsigned short* gB[8];
  unsigned lOff[8];
#pragma unroll
  for (int t = 0; t < 8; ++t) {
    int id  = t * 32 + lane;
    int row = id >> 2;            // 0..63  (C-column within tile)
    int ko  = (id & 3) * 8;       // 0,8,16,24
    gB[t]   = B + (size_t)(n0 + row) * K + ko;
    // low 32 bits of a flat LDS pointer are the LDS byte offset
    lOff[t] = (unsigned)(size_t)(Bs + row * BS_STRIDE + ko);
  }

  for (int k0 = 0; k0 < K; k0 += 32) {
    // async-copy B slab into LDS (tracked by ASYNCcnt)
#pragma unroll
    for (int t = 0; t < 8; ++t)
      asm volatile("global_load_async_to_lds_b128 %0, %1, off"
                   :: "v"(lOff[t]), "v"(gB[t] + k0)
                   : "memory");

    // A fragments direct from global (overlaps with the async copy)
    v16bf a[4];
#pragma unroll
    for (int i = 0; i < 4; ++i)
      a[i] = load_a_frag(A, m0 + i * 16 + lc, K, k0, lane);

    if (k0 + 32 < K)
      __builtin_prefetch(A + (size_t)(m0 + lc) * K + k0 + 32, 0, 1);

    asm volatile("s_wait_asynccnt 0x0" ::: "memory");

#pragma unroll
    for (int j = 0; j < 4; ++j) {
      const unsigned short* p = Bs + (j * 16 + lc) * BS_STRIDE + ((lane >> 4) << 4);
      Frag f;
      f.q[0] = *(const v4u*)p;
      f.q[1] = *(const v4u*)(p + 8);
#pragma unroll
      for (int i = 0; i < 4; ++i)
        acc[i][j] = WMMA_BF16(a[i], f.v, acc[i][j]);
    }
    // safe single-buffer reuse: the s_wait_dscnt before each WMMA retires all
    // LDS reads before the next iteration's async stores are issued.
  }
}

// ---------------- generic GEMM, fp32 C (output projection) ----------------
__global__ __launch_bounds__(32) void gemm_bf16_nt(const unsigned short* __restrict__ A,
                                                   const unsigned short* __restrict__ B,
                                                   float* __restrict__ C,
                                                   int M, int N, int K) {
  __shared__ __align__(16) unsigned short Bs[64 * BS_STRIDE];
  const int lane = threadIdx.x & 31;
  const int m0 = blockIdx.x * 64;
  const int n0 = blockIdx.y * 64;

  v8f acc[4][4] = {};
  gemm_mainloop(A, B, Bs, m0, n0, K, lane, acc);

  const int rbase = (lane >> 4) * 8;
  const int lc = lane & 15;
#pragma unroll
  for (int i = 0; i < 4; ++i)
#pragma unroll
    for (int j = 0; j < 4; ++j) {
      int row = m0 + i * 16 + rbase;
      int col = n0 + j * 16 + lc;
#pragma unroll
      for (int r = 0; r < 8; ++r)
        C[(size_t)(row + r) * N + col] = acc[i][j][r];
    }
}

// ---------------- QKV GEMM with fused RoPE / GQA split / V-transpose ----------------
// Each 64-wide N-tile is exactly one head, so the RoPE pair (d, d+/-32) lives in
// accumulator tiles j and j+/-2 of the SAME wave/lane/row: fuse it in registers.
//   heads  0..31 -> qb  bf16 [N_HEADS][S][64]  (RoPE'd, pre-scaled 1/sqrt(64))
//   heads 32..39 -> kb  bf16 [N_KV][S][64]     (RoPE'd)
//   heads 40..47 -> vtb bf16 [N_KV][64][S]     (transposed for P*V B-fragments)
__global__ __launch_bounds__(32) void gemm_qkv_rope(const unsigned short* __restrict__ A,
                                                    const unsigned short* __restrict__ B,
                                                    const float* __restrict__ cosb,
                                                    const float* __restrict__ sinb,
                                                    unsigned short* __restrict__ qb,
                                                    unsigned short* __restrict__ kb,
                                                    unsigned short* __restrict__ vtb) {
  __shared__ __align__(16) unsigned short Bs[64 * BS_STRIDE];
  const int lane = threadIdx.x & 31;
  const int m0 = blockIdx.x * 64;
  const int hh = blockIdx.y;          // fused head index 0..47
  const int n0 = hh * 64;

  v8f acc[4][4] = {};
  gemm_mainloop(A, B, Bs, m0, n0, HIDDEN, lane, acc);

  const int rbase = (lane >> 4) * 8;
  const int lc = lane & 15;

  if (hh < N_HEADS + N_KV) {
    // Q or K head: apply RoPE using the in-register (d, d+/-32) pair
#pragma unroll
    for (int i = 0; i < 4; ++i)
#pragma unroll
      for (int r = 0; r < 8; ++r) {
        const int row = m0 + i * 16 + rbase + r;      // sequence position s
#pragma unroll
        for (int j = 0; j < 4; ++j) {
          const int d = j * 16 + lc;
          float c  = cosb[row * HEAD_D + d];
          float sn = sinb[row * HEAD_D + d];
          float val   = acc[i][j][r];
          float other = (j < 2) ? -acc[i][j + 2][r] : acc[i][j - 2][r];
          float res = val * c + other * sn;
          if (hh < N_HEADS)
            qb[((size_t)hh * S_LEN + row) * HEAD_D + d] = f2bf(res * 0.125f);
          else
            kb[((size_t)(hh - N_HEADS) * S_LEN + row) * HEAD_D + d] = f2bf(res);
        }
      }
  } else {
    // V head: no RoPE, store transposed [d][s]
    const int vh = hh - N_HEADS - N_KV;
#pragma unroll
    for (int i = 0; i < 4; ++i)
#pragma unroll
      for (int j = 0; j < 4; ++j) {
        const int d = j * 16 + lc;
#pragma unroll
        for (int r = 0; r < 8; ++r) {
          const int row = m0 + i * 16 + rbase + r;
          vtb[((size_t)vh * HEAD_D + d) * S_LEN + row] = f2bf(acc[i][j][r]);
        }
      }
  }
}

// ---------------- flash attention (causal, GQA 4:1) ----------------
// Block = 128 threads = 4 waves; each wave owns 32 query rows of one head.
#define P_STRIDE 40   // bf16 elems per staged P row (80B, keeps 16B alignment)

__global__ __launch_bounds__(128) void attn_fa(const unsigned short* __restrict__ qb,
                                               const unsigned short* __restrict__ kb,
                                               const unsigned short* __restrict__ vtb,
                                               unsigned short* __restrict__ aob) {
  __shared__ __align__(16) unsigned short pshm[4][32 * P_STRIDE];

  const int lane = threadIdx.x & 31;
  const int warp = threadIdx.x >> 5;
  const int half = lane >> 4;     // 0 or 1
  const int lc   = lane & 15;

  const int h   = blockIdx.y;
  const int kvh = h >> 2;         // GQA group of 4
  const int q0  = blockIdx.x * 128 + warp * 32;

  const unsigned short* Q  = qb  + (size_t)h   * S_LEN * HEAD_D;
  const unsigned short* Kp = kb  + (size_t)kvh * S_LEN * HEAD_D;
  const unsigned short* Vt = vtb + (size_t)kvh * HEAD_D * S_LEN;
  unsigned short* Pl = &pshm[warp][0];

  // Q fragments: 2 row-subtiles x 2 K-chunks (head dim 64 = 2x32)
  v16bf qf[2][2];
#pragma unroll
  for (int i = 0; i < 2; ++i)
#pragma unroll
    for (int t = 0; t < 2; ++t)
      qf[i][t] = load_a_frag(Q, q0 + i * 16 + lc, HEAD_D, t * 32, lane);

  v8f oacc[2][4] = {};
  float mrun[2][8], lrun[2][8];
#pragma unroll
  for (int i = 0; i < 2; ++i)
#pragma unroll
    for (int r = 0; r < 8; ++r) { mrun[i][r] = -1e30f; lrun[i][r] = 0.f; }

  for (int k0 = 0; k0 <= q0; k0 += 32) {
    // ---- S = Q * K^T over a 32-key slab ----
    v8f sacc[2][2] = {};
#pragma unroll
    for (int n = 0; n < 2; ++n) {
      v16bf kf0 = load_b_frag(Kp, k0 + n * 16 + lc, HEAD_D, 0,  lane);
      v16bf kf1 = load_b_frag(Kp, k0 + n * 16 + lc, HEAD_D, 32, lane);
#pragma unroll
      for (int i = 0; i < 2; ++i) {
        sacc[i][n] = WMMA_BF16(qf[i][0], kf0, sacc[i][n]);
        sacc[i][n] = WMMA_BF16(qf[i][1], kf1, sacc[i][n]);
      }
    }

    const bool diag = (k0 == q0);

    // ---- online softmax (fp32, 16-lane row reductions per half-wave) ----
#pragma unroll
    for (int i = 0; i < 2; ++i) {
#pragma unroll
      for (int r = 0; r < 8; ++r) {
        const int row = q0 + i * 16 + half * 8 + r;
        float s0 = sacc[i][0][r];
        float s1 = sacc[i][1][r];
        if (diag) {
          if (k0 + lc      > row) s0 = -1e30f;
          if (k0 + 16 + lc > row) s1 = -1e30f;
        }
        float cm = fmaxf(s0, s1);
#pragma unroll
        for (int off = 1; off < 16; off <<= 1)
          cm = fmaxf(cm, __shfl_xor(cm, off, 32));
        float mnew  = fmaxf(mrun[i][r], cm);
        float alpha = __expf(mrun[i][r] - mnew);
        float p0 = __expf(s0 - mnew);
        float p1 = __expf(s1 - mnew);
        float ls = p0 + p1;
#pragma unroll
        for (int off = 1; off < 16; off <<= 1)
          ls += __shfl_xor(ls, off, 32);
        lrun[i][r] = lrun[i][r] * alpha + ls;
        mrun[i][r] = mnew;
#pragma unroll
        for (int jv = 0; jv < 4; ++jv) oacc[i][jv][r] *= alpha;

        // stage P (C-layout -> row-major LDS) for the A-fragment re-read
        Pl[(i * 16 + half * 8 + r) * P_STRIDE + lc]      = f2bf(p0);
        Pl[(i * 16 + half * 8 + r) * P_STRIDE + 16 + lc] = f2bf(p1);
      }
    }

    // all ds_stores of this wave must land before layout-changing re-read
    asm volatile("s_wait_dscnt 0x0" ::: "memory");

    // ---- re-read P as A-fragments (16x32 bf16 each) ----
    v16bf pf[2];
#pragma unroll
    for (int i = 0; i < 2; ++i) {
      const unsigned short* pp = Pl + (i * 16 + lc) * P_STRIDE + half * 16;
      Frag f;
      f.q[0] = *(const v4u*)pp;
      f.q[1] = *(const v4u*)(pp + 8);
      pf[i] = f.v;
    }

    // ---- O += P * V  (V transposed: per-lane K-contiguous B-fragments) ----
#pragma unroll
    for (int jv = 0; jv < 4; ++jv) {
      const unsigned short* vp = Vt + (size_t)(jv * 16 + lc) * S_LEN + k0 + half * 16;
      Frag f;
      f.q[0] = *(const v4u*)vp;
      f.q[1] = *(const v4u*)(vp + 8);
      v16bf vf = f.v;
#pragma unroll
      for (int i = 0; i < 2; ++i)
        oacc[i][jv] = WMMA_BF16(pf[i], vf, oacc[i][jv]);
    }
  }

  // ---- normalize and write bf16 attn output [S][N_HEADS*64] ----
#pragma unroll
  for (int i = 0; i < 2; ++i)
#pragma unroll
    for (int r = 0; r < 8; ++r) {
      const float inv = 1.0f / lrun[i][r];
      const int row = q0 + i * 16 + half * 8 + r;
#pragma unroll
      for (int jv = 0; jv < 4; ++jv)
        aob[(size_t)row * AO_COLS + h * HEAD_D + jv * 16 + lc] =
            f2bf(oacc[i][jv][r] * inv);
    }
}

// ---------------- host launcher ----------------
extern "C" void kernel_launch(void* const* d_in, const int* in_sizes, int n_in,
                              void* d_out, int out_size, void* d_ws, size_t ws_size,
                              hipStream_t stream) {
  const float* x    = (const float*)d_in[0];   // [S][HIDDEN]
  const float* cosb = (const float*)d_in[1];   // [S][64]
  const float* sinb = (const float*)d_in[2];   // [S][64]
  const float* wqkv = (const float*)d_in[3];   // [3072][HIDDEN]
  const float* wout = (const float*)d_in[4];   // [HIDDEN][2048]
  float* out = (float*)d_out;                  // [S][HIDDEN]

  char* ws = (char*)d_ws;
  size_t off = 0;
  auto carve = [&](size_t bytes) -> char* {
    char* p = ws + off;
    off += (bytes + 255) & ~(size_t)255;
    return p;
  };
  unsigned short* xb    = (unsigned short*)carve((size_t)S_LEN * HIDDEN * 2);           // 8 MB
  unsigned short* wqkvb = (unsigned short*)carve((size_t)QKVO * HIDDEN * 2);            // 12 MB
  unsigned short* woutb = (unsigned short*)carve((size_t)HIDDEN * AO_COLS * 2);         // 8 MB
  unsigned short* qb    = (unsigned short*)carve((size_t)N_HEADS * S_LEN * HEAD_D * 2); // 8 MB
  unsigned short* kb    = (unsigned short*)carve((size_t)N_KV * S_LEN * HEAD_D * 2);    // 2 MB
  unsigned short* vtb   = (unsigned short*)carve((size_t)N_KV * HEAD_D * S_LEN * 2);    // 2 MB
  unsigned short* aob   = (unsigned short*)carve((size_t)S_LEN * AO_COLS * 2);          // 8 MB
  (void)ws_size;

  // 1) fp32 -> bf16 for the three GEMM operands
  cvt_bf16<<<2048, 256, 0, stream>>>(x,    xb,    S_LEN * HIDDEN);
  cvt_bf16<<<2048, 256, 0, stream>>>(wqkv, wqkvb, QKVO * HIDDEN);
  cvt_bf16<<<2048, 256, 0, stream>>>(wout, woutb, HIDDEN * AO_COLS);

  // 2) QKV projection with fused RoPE / GQA split / V-transpose
  gemm_qkv_rope<<<dim3(S_LEN / 64, QKVO / 64), 32, 0, stream>>>(
      xb, wqkvb, cosb, sinb, qb, kb, vtb);

  // 3) causal GQA flash attention
  attn_fa<<<dim3(S_LEN / 128, N_HEADS), 128, 0, stream>>>(qb, kb, vtb, aob);

  // 4) output projection: out[s,h'] = sum_o attn[s,o] * w_out[h',o]
  gemm_bf16_nt<<<dim3(S_LEN / 64, HIDDEN / 64), 32, 0, stream>>>(
      aob, woutb, out, S_LEN, HIDDEN, AO_COLS);
}
